// ScaledRoPEAttention_85822036508880
// MI455X (gfx1250) — matrix operand order repaired and checked by reading, not compiled, not run on hardware
//
#include <hip/hip_runtime.h>

// ---------------------------------------------------------------------------
// ScaledRoPEAttention for MI455X (gfx1250): bf16 WMMA pipeline
//   x --(GEMM)--> Q,K,V[B,H,S,D] --(RoPE)--> flash-attn --> ctx --(GEMM)--> out
// GEMMs: 128x128 tile / 256 threads, double-buffered LDS fed by
// global_load_async_to_lds_b128 (ASYNCcnt). All matrix math:
// v_wmma_f32_16x16x32_bf16 with fp32 accumulation.
// ---------------------------------------------------------------------------

typedef __bf16 bf16;
typedef __attribute__((ext_vector_type(16))) __bf16 v16bf;
typedef __attribute__((ext_vector_type(8)))  float  v8f;

#define EMBED 2048
#define NHEAD 16
#define HDIM  128
#define BATCH 2
#define SEQ   2048
#define MTOT  (BATCH * SEQ)   // 4096 rows of x / ctx

__device__ __forceinline__ bf16 f2bf(float f) {
  unsigned u = __builtin_bit_cast(unsigned, f);
  unsigned r = (u + 0x7FFFu + ((u >> 16) & 1u)) >> 16;   // round-to-nearest-even
  return __builtin_bit_cast(bf16, (unsigned short)r);
}
__device__ __forceinline__ float bf2f(bf16 b) {
  unsigned u = ((unsigned)__builtin_bit_cast(unsigned short, b)) << 16;
  return __builtin_bit_cast(float, u);
}

// LDS byte address of a __shared__ location: ISA aperture rule -> low 32 bits
// of the generic address are the wave-relative LDS offset.
__device__ __forceinline__ unsigned ldsaddr(const void* p) {
  return (unsigned)(size_t)p;
}
// Direct global -> LDS async copy, 16B per lane (tracked by ASYNCcnt).
__device__ __forceinline__ void async_copy_b128(unsigned lds, const bf16* g) {
  asm volatile("global_load_async_to_lds_b128 %0, %1, off"
               :: "v"(lds), "v"(g) : "memory");
}
__device__ __forceinline__ void wait_async0() {
  asm volatile("s_wait_asynccnt 0x0" ::: "memory");
}

// ------------------------------ fp32 -> bf16 -------------------------------
__global__ void cvt_f32_bf16_kernel(const float* __restrict__ in,
                                    bf16* __restrict__ out, int n) {
  for (int i = blockIdx.x * blockDim.x + threadIdx.x; i < n;
       i += gridDim.x * blockDim.x)
    out[i] = f2bf(in[i]);
}

// ------------------------- NT-GEMM main loop (LDS) -------------------------
// Workgroup: 256 threads = 8 waves; output tile 128(M) x 128(N).
// wave w: m-slice = (w>>1)*32 (two 16-row subtiles), n-slice = (w&1)*64
// (four 16-wide n-subtiles) -> 8 WMMA per K-step of 32.
// A[m,k] row-major (x/ctx), Bw[n,k] row-major (torch Linear: y = x W^T).
__device__ __forceinline__ void gemm_tile_mainloop(
    const bf16* __restrict__ A, const bf16* __restrict__ Bw,
    int m0, int n0, v8f acc[2][4]) {
  __shared__ __align__(16) bf16 At[2][128 * 32];
  __shared__ __align__(16) bf16 Bt[2][128 * 32];
  const int tid  = threadIdx.x;
  const int lane = tid & 31;
  const int wave = tid >> 5;
  const int msl  = (wave >> 1) * 32;
  const int nsl  = (wave & 1) * 64;
  const int l15  = lane & 15, lg = lane >> 4;

#pragma unroll
  for (int mi = 0; mi < 2; mi++)
#pragma unroll
    for (int nt = 0; nt < 4; nt++) acc[mi][nt] = (v8f){0, 0, 0, 0, 0, 0, 0, 0};

  const int lr = tid >> 2;        // 0..63
  const int lc = (tid & 3) * 8;   // 0,8,16,24

  // issue async loads for one 128x32 A tile + 128x32 B tile into buffer `buf`
  auto issue = [&](int buf, int k0) {
    async_copy_b128(ldsaddr(&At[buf][lr * 32 + lc]),
                    &A[(size_t)(m0 + lr) * EMBED + k0 + lc]);
    async_copy_b128(ldsaddr(&At[buf][(64 + lr) * 32 + lc]),
                    &A[(size_t)(m0 + 64 + lr) * EMBED + k0 + lc]);
    async_copy_b128(ldsaddr(&Bt[buf][lr * 32 + lc]),
                    &Bw[(size_t)(n0 + lr) * EMBED + k0 + lc]);
    async_copy_b128(ldsaddr(&Bt[buf][(64 + lr) * 32 + lc]),
                    &Bw[(size_t)(n0 + 64 + lr) * EMBED + k0 + lc]);
  };

  issue(0, 0);
  wait_async0();
  __syncthreads();

  const int NIT = EMBED / 32;
  for (int it = 0; it < NIT; ++it) {
    const int cur = it & 1;
    if (it + 1 < NIT) issue(cur ^ 1, (it + 1) * 32);   // prefetch next tile

    // A fragments: lane holds row (lane&15);
    // elem e -> k = lg*16+e
    v16bf a[2];
#pragma unroll
    for (int mi = 0; mi < 2; mi++) {
      const bf16* ap = &At[cur][(msl + mi * 16 + l15) * 32];
#pragma unroll
      for (int e = 0; e < 8; e++) a[mi][e] = ap[lg * 16 + e];
#pragma unroll
      for (int e = 0; e < 8; e++) a[mi][8 + e] = ap[lg * 16 + 8 + e];
    }
#pragma unroll
    for (int nt = 0; nt < 4; nt++) {
      // B fragment: elem e of lane l = B[lg*16+e][l&15]; B[k][n] = W[n][k]
      v16bf bm;
      const bf16* bp = &Bt[cur][(nsl + nt * 16 + l15) * 32];
#pragma unroll
      for (int e = 0; e < 16; e++) bm[e] = bp[lg * 16 + e];
      acc[0][nt] = __builtin_amdgcn_wmma_f32_16x16x32_bf16(
          false, a[0], false, bm, (short)0, acc[0][nt], false, false);
      acc[1][nt] = __builtin_amdgcn_wmma_f32_16x16x32_bf16(
          false, a[1], false, bm, (short)0, acc[1][nt], false, false);
    }

    if (it + 1 < NIT) {
      wait_async0();     // next tile landed in LDS
      __syncthreads();   // all waves done reading `cur`, next buffer visible
    }
  }
}

// ------------------- GEMM + bias, scatter to [B,H,S,D] ---------------------
__global__ void gemm_qkv_kernel(const bf16* __restrict__ xb,
                                const bf16* __restrict__ Wb,
                                const float* __restrict__ bias,
                                bf16* __restrict__ out) {
  const int m0 = blockIdx.x * 128;
  const int n0 = blockIdx.y * 128;
  v8f acc[2][4];
  gemm_tile_mainloop(xb, Wb, m0, n0, acc);

  const int lane = threadIdx.x & 31, wave = threadIdx.x >> 5;
  const int msl = (wave >> 1) * 32, nsl = (wave & 1) * 64;
  const int l15 = lane & 15, lg = lane >> 4;
#pragma unroll
  for (int mi = 0; mi < 2; mi++) {
#pragma unroll
    for (int nt = 0; nt < 4; nt++) {
      const int n  = n0 + nsl + nt * 16 + l15;
      const int h  = n >> 7, d = n & 127;
      const float bv = bias[n];
#pragma unroll
      for (int rr = 0; rr < 8; rr++) {
        const int m = m0 + msl + mi * 16 + lg * 8 + rr;  // C-layout row
        const int b = m >> 11, s = m & (SEQ - 1);
        out[(((size_t)(b * NHEAD + h)) * SEQ + s) * HDIM + d] =
            f2bf(acc[mi][nt][rr] + bv);
      }
    }
  }
}

// ------------------- GEMM + bias, fp32 output [M,E] ------------------------
__global__ void gemm_out_kernel(const bf16* __restrict__ ctx,
                                const bf16* __restrict__ Wb,
                                const float* __restrict__ bias,
                                float* __restrict__ out) {
  const int m0 = blockIdx.x * 128;
  const int n0 = blockIdx.y * 128;
  v8f acc[2][4];
  gemm_tile_mainloop(ctx, Wb, m0, n0, acc);

  const int lane = threadIdx.x & 31, wave = threadIdx.x >> 5;
  const int msl = (wave >> 1) * 32, nsl = (wave & 1) * 64;
  const int l15 = lane & 15, lg = lane >> 4;
#pragma unroll
  for (int mi = 0; mi < 2; mi++) {
#pragma unroll
    for (int nt = 0; nt < 4; nt++) {
      const int n  = n0 + nsl + nt * 16 + l15;
      const float bv = bias[n];
#pragma unroll
      for (int rr = 0; rr < 8; rr++) {
        const int m = m0 + msl + mi * 16 + lg * 8 + rr;
        out[(size_t)m * EMBED + n] = acc[mi][nt][rr] + bv;
      }
    }
  }
}

// --------------------------- RoPE (half-split) -----------------------------
__global__ void rope_kernel(bf16* __restrict__ t) {
  const int NP = BATCH * NHEAD * SEQ * (HDIM / 2);  // pairs
  int i = blockIdx.x * blockDim.x + threadIdx.x;
  if (i >= NP) return;
  const int j  = i & 63;
  const int s  = (i >> 6) & (SEQ - 1);
  const int bh = i >> 17;
  const size_t base = ((size_t)bh * SEQ + s) * HDIM;
  const float inv = __expf(-__logf(10000.f) * ((float)(2 * j) / (float)HDIM));
  const float ang = (float)s * inv;
  float sn, cs;
  __sincosf(ang, &sn, &cs);
  const float x0 = bf2f(t[base + j]);
  const float x1 = bf2f(t[base + 64 + j]);
  t[base + j]      = f2bf(x0 * cs - x1 * sn);
  t[base + 64 + j] = f2bf(x1 * cs + x0 * sn);
}

// --------------------- Flash attention (online softmax) --------------------
// grid: (S/64, B*H); block: 128 threads = 4 waves; each wave owns 16 q-rows.
// Streams K-blocks of 32; scores and PV via bf16 WMMA; never materializes SxS.
// K tile staged with async-to-LDS; V transposed during staging.
__global__ void flash_attn_kernel(const bf16* __restrict__ Q,
                                  const bf16* __restrict__ K,
                                  const bf16* __restrict__ V,
                                  bf16* __restrict__ ctx) {
  const int qb  = blockIdx.x;
  const int bh  = blockIdx.y;
  const int b   = bh >> 4;
  const int h   = bh & 15;
  const int tid = threadIdx.x;
  const int lane = tid & 31, wave = tid >> 5;
  const int l15 = lane & 15, lg = lane >> 4;

  __shared__ __align__(16) bf16 Kt[32 * 128];   // K rows t0..t0+31, full D
  __shared__ __align__(16) bf16 VTt[128 * 32];  // V transposed: [d][t]
  __shared__ __align__(16) bf16 Pt[4][16 * 32]; // per-wave P staging

  const size_t qkbase = (size_t)bh * SEQ * HDIM;
  const int mrow0 = qb * 64 + wave * 16;

  // resident Q fragments: 4 d-chunks of K=32 (A-fragment layout)
  v16bf qf[4];
  {
    const bf16* qp = &Q[qkbase + (size_t)(mrow0 + l15) * HDIM];
#pragma unroll
    for (int dc = 0; dc < 4; dc++) {
#pragma unroll
      for (int e = 0; e < 8; e++) qf[dc][e] = qp[dc * 32 + lg * 16 + e];
#pragma unroll
      for (int e = 0; e < 8; e++) qf[dc][8 + e] = qp[dc * 32 + lg * 16 + 8 + e];
    }
  }

  v8f oacc[8];
#pragma unroll
  for (int i = 0; i < 8; i++) oacc[i] = (v8f){0, 0, 0, 0, 0, 0, 0, 0};
  float mrun[8], lrun[8];
#pragma unroll
  for (int rr = 0; rr < 8; rr++) { mrun[rr] = -1e30f; lrun[rr] = 0.f; }

  const float scale = 0.08838834764831845f;  // 1/sqrt(128)

  const int ldr = tid >> 2;         // 0..31 : K/V row within block
  const int ldc = (tid & 3) * 32;   // 0,32,64,96

  for (int t0 = 0; t0 < SEQ; t0 += 32) {
    __syncthreads();
    // K tile: async global->LDS (4 x 16B per thread); V: transpose via VGPRs
    {
      const bf16* krow = &K[qkbase + (size_t)(t0 + ldr) * HDIM + ldc];
      async_copy_b128(ldsaddr(&Kt[ldr * 128 + ldc]),      krow);
      async_copy_b128(ldsaddr(&Kt[ldr * 128 + ldc + 8]),  krow + 8);
      async_copy_b128(ldsaddr(&Kt[ldr * 128 + ldc + 16]), krow + 16);
      async_copy_b128(ldsaddr(&Kt[ldr * 128 + ldc + 24]), krow + 24);
      const bf16* vrow = &V[qkbase + (size_t)(t0 + ldr) * HDIM + ldc];
#pragma unroll
      for (int e = 0; e < 32; e++) VTt[(ldc + e) * 32 + ldr] = vrow[e];
    }
    wait_async0();
    __syncthreads();

    // scores: S = Q (16xD) x K^T; two 16-wide t-chunks
    v8f s0 = (v8f){0, 0, 0, 0, 0, 0, 0, 0};
    v8f s1 = (v8f){0, 0, 0, 0, 0, 0, 0, 0};
#pragma unroll
    for (int dc = 0; dc < 4; dc++) {
      v16bf kb0, kb1;
      const bf16* kp0 = &Kt[l15 * 128 + dc * 32];
      const bf16* kp1 = &Kt[(16 + l15) * 128 + dc * 32];
#pragma unroll
      for (int e = 0; e < 16; e++) {
        kb0[e] = kp0[lg * 16 + e];
        kb1[e] = kp1[lg * 16 + e];
      }
      s0 = __builtin_amdgcn_wmma_f32_16x16x32_bf16(false, qf[dc], false, kb0,
                                                   (short)0, s0, false, false);
      s1 = __builtin_amdgcn_wmma_f32_16x16x32_bf16(false, qf[dc], false, kb1,
                                                   (short)0, s1, false, false);
    }

    // online softmax: row r lives in the 16-lane group sharing (lane>>4)
    float p0[8], p1[8];
#pragma unroll
    for (int rr = 0; rr < 8; rr++) {
      float sa = s0[rr] * scale, sb = s1[rr] * scale;
      float mx = fmaxf(sa, sb);
      mx = fmaxf(mx, __shfl_xor(mx, 1, 32));
      mx = fmaxf(mx, __shfl_xor(mx, 2, 32));
      mx = fmaxf(mx, __shfl_xor(mx, 4, 32));
      mx = fmaxf(mx, __shfl_xor(mx, 8, 32));
      const float mnew  = fmaxf(mrun[rr], mx);
      const float alpha = __expf(mrun[rr] - mnew);
      const float e0 = __expf(sa - mnew);
      const float e1 = __expf(sb - mnew);
      float rs = e0 + e1;
      rs += __shfl_xor(rs, 1, 32);
      rs += __shfl_xor(rs, 2, 32);
      rs += __shfl_xor(rs, 4, 32);
      rs += __shfl_xor(rs, 8, 32);
      lrun[rr] = lrun[rr] * alpha + rs;
      mrun[rr] = mnew;
      p0[rr] = e0;
      p1[rr] = e1;
#pragma unroll
      for (int dc = 0; dc < 8; dc++) oacc[dc][rr] *= alpha;
    }

    // C-layout -> A-fragment: stage P through LDS
#pragma unroll
    for (int rr = 0; rr < 8; rr++) {
      Pt[wave][(lg * 8 + rr) * 32 + l15]      = f2bf(p0[rr]);
      Pt[wave][(lg * 8 + rr) * 32 + 16 + l15] = f2bf(p1[rr]);
    }
    __syncthreads();

    v16bf pf;
    {
      const bf16* pp = &Pt[wave][l15 * 32];
#pragma unroll
      for (int e = 0; e < 8; e++) pf[e] = pp[lg * 16 + e];
#pragma unroll
      for (int e = 0; e < 8; e++) pf[8 + e] = pp[lg * 16 + 8 + e];
    }
    // O += P (16x32) x V (32xD)
#pragma unroll
    for (int dc = 0; dc < 8; dc++) {
      v16bf vb;
      const bf16* vp = &VTt[(dc * 16 + l15) * 32 + lg * 16];
#pragma unroll
      for (int e = 0; e < 16; e++) vb[e] = vp[e];
      oacc[dc] = __builtin_amdgcn_wmma_f32_16x16x32_bf16(
          false, pf, false, vb, (short)0, oacc[dc], false, false);
    }
  }

  // epilogue: normalize and write ctx as [B,S,E] bf16 (E = h*128 + d)
#pragma unroll
  for (int rr = 0; rr < 8; rr++) {
    const float inv = 1.f / lrun[rr];
    const int srow  = mrow0 + lg * 8 + rr;
    const size_t ob = ((size_t)b * SEQ + srow) * EMBED + h * HDIM;
#pragma unroll
    for (int dc = 0; dc < 8; dc++)
      ctx[ob + dc * 16 + l15] = f2bf(oacc[dc][rr] * inv);
  }
}

// ---------------------------------------------------------------------------
extern "C" void kernel_launch(void* const* d_in, const int* in_sizes, int n_in,
                              void* d_out, int out_size, void* d_ws,
                              size_t ws_size, hipStream_t stream) {
  const float* x  = (const float*)d_in[0];
  const float* Wq = (const float*)d_in[1];
  const float* bq = (const float*)d_in[2];
  const float* Wk = (const float*)d_in[3];
  const float* bk = (const float*)d_in[4];
  const float* Wv = (const float*)d_in[5];
  const float* bv = (const float*)d_in[6];
  const float* Wo = (const float*)d_in[7];
  const float* bo = (const float*)d_in[8];
  float* out = (float*)d_out;

  const size_t xbytes = (size_t)MTOT * EMBED * sizeof(bf16);       // 16 MiB
  const size_t wbytes = (size_t)EMBED * EMBED * sizeof(bf16);      // 8 MiB
  const size_t qbytes = (size_t)BATCH * NHEAD * SEQ * HDIM * sizeof(bf16);

  char* p = (char*)d_ws;
  bf16* xb   = (bf16*)p; p += xbytes;
  bf16* Wqb  = (bf16*)p; p += wbytes;
  bf16* Wkb  = (bf16*)p; p += wbytes;
  bf16* Wvb  = (bf16*)p; p += wbytes;
  bf16* Wob  = (bf16*)p; p += wbytes;
  bf16* Qb   = (bf16*)p; p += qbytes;
  bf16* Kb   = (bf16*)p; p += qbytes;
  bf16* Vb   = (bf16*)p; p += qbytes;
  bf16* Ctxb = (bf16*)p; p += xbytes;   // ~112 MiB total

  // 1) fp32 -> bf16
  cvt_f32_bf16_kernel<<<8192, 256, 0, stream>>>(x, xb, MTOT * EMBED);
  cvt_f32_bf16_kernel<<<8192, 256, 0, stream>>>(Wq, Wqb, EMBED * EMBED);
  cvt_f32_bf16_kernel<<<8192, 256, 0, stream>>>(Wk, Wkb, EMBED * EMBED);
  cvt_f32_bf16_kernel<<<8192, 256, 0, stream>>>(Wv, Wvb, EMBED * EMBED);
  cvt_f32_bf16_kernel<<<8192, 256, 0, stream>>>(Wo, Wob, EMBED * EMBED);

  // 2) QKV projections (scatter to [B,H,S,D])
  dim3 ggrid(MTOT / 128, EMBED / 128);
  gemm_qkv_kernel<<<ggrid, 256, 0, stream>>>(xb, Wqb, bq, Qb);
  gemm_qkv_kernel<<<ggrid, 256, 0, stream>>>(xb, Wkb, bk, Kb);
  gemm_qkv_kernel<<<ggrid, 256, 0, stream>>>(xb, Wvb, bv, Vb);

  // 3) RoPE on Q and K
  const int npairs = BATCH * NHEAD * SEQ * (HDIM / 2);
  rope_kernel<<<npairs / 256, 256, 0, stream>>>(Qb);
  rope_kernel<<<npairs / 256, 256, 0, stream>>>(Kb);

  // 4) flash attention -> ctx [B,S,E] bf16
  flash_attn_kernel<<<dim3(SEQ / 64, BATCH * NHEAD), 128, 0, stream>>>(
      Qb, Kb, Vb, Ctxb);

  // 5) output projection -> fp32
  gemm_out_kernel<<<ggrid, 256, 0, stream>>>(Ctxb, Wob, bo, out);
}
